// LongformerMultiHeadAttention_74826920230969
// MI455X (gfx1250) — matrix operand order repaired and checked by reading, not compiled, not run on hardware
//
#include <hip/hip_runtime.h>

// ---------------------------------------------------------------------------
// Longformer MHA for MI455X (gfx1250):
//  - all contractions on v_wmma_f32_16x16x32_f16 (f32 accumulate)
//  - global->LDS staging via global_load_async_to_lds_b128 (ASYNCcnt),
//    double-buffered in the GEMM so WMMA overlaps the memory pipe
// B=2, S=4096, D=1024, H=16, DH=64, W1=128, C=32 chunks, G=64, window=257
// ---------------------------------------------------------------------------

typedef __attribute__((ext_vector_type(16))) _Float16 v16h;
typedef __attribute__((ext_vector_type(8)))  float    v8f;

union F16Frag { v16h v; uint4 u[2]; };

#define WMMA(A, B, C) \
  __builtin_amdgcn_wmma_f32_16x16x32_f16(false, (A), false, (B), (short)0, (C), false, false)

#define NSEQ 4096
#define NDIM 1024
#define NHEAD 16
#define NDH 64
#define NEGV (-1.0e9f)

// 16-byte async copy global -> LDS (per active lane). LDS byte address is the
// low 32 bits of the generic shared-memory pointer.
__device__ __forceinline__ void async_cp16(unsigned ldsAddr, const void* gptr) {
  asm volatile("global_load_async_to_lds_b128 %0, %1, off"
               :: "v"(ldsAddr), "v"((unsigned long long)(uintptr_t)gptr)
               : "memory");
}
__device__ __forceinline__ unsigned lds_addr(const void* p) {
  return (unsigned)(uintptr_t)p;
}

// ----------------------------- converters ---------------------------------

__global__ __launch_bounds__(256) void cvt_f32_to_f16(
    const float* __restrict__ in, _Float16* __restrict__ out, int n4) {
  int i = blockIdx.x * blockDim.x + threadIdx.x;
  if (i < n4) {
    float4 f = ((const float4*)in)[i];
    union { _Float16 h[4]; uint2 u; } pk;
    pk.h[0] = (_Float16)f.x; pk.h[1] = (_Float16)f.y;
    pk.h[2] = (_Float16)f.z; pk.h[3] = (_Float16)f.w;
    ((uint2*)out)[i] = pk.u;
  }
}

// W[k][n] f32  ->  Wt[n][k] f16   (1024x1024)
__global__ __launch_bounds__(256) void cvt_transpose(
    const float* __restrict__ W, _Float16* __restrict__ Wt) {
  __shared__ float tb[32][33];
  int bx = blockIdx.x * 32, by = blockIdx.y * 32;
  int tx = threadIdx.x & 31, ty = threadIdx.x >> 5;   // 32 x 8
  for (int i = 0; i < 32; i += 8)
    tb[ty + i][tx] = W[(size_t)(by + ty + i) * NDIM + bx + tx];
  __syncthreads();
  for (int i = 0; i < 32; i += 8)
    Wt[(size_t)(bx + ty + i) * NDIM + by + tx] = (_Float16)tb[tx][ty + i];
}

// ------------------------------- GEMM --------------------------------------
// C[8192x1024] = A[8192x1024] * Wt^T  (Wt is n-major f16)
// mode 0/1/2: store f16 into [b][h][s][dh] buffers (mode 0 scaled by 1/8)
// mode 3    : store f32 row-major into Dout
// 128x128 block tile, K-step 32, double-buffered async global->LDS staging.
__global__ __launch_bounds__(256) void gemm128(
    const _Float16* __restrict__ A,
    const _Float16* __restrict__ Wt0, const _Float16* __restrict__ Wt1,
    const _Float16* __restrict__ Wt2,
    _Float16* __restrict__ D0, _Float16* __restrict__ D1,
    _Float16* __restrict__ D2, float* __restrict__ Dout, int modeBase) {
  __shared__ __align__(16) _Float16 As[2][128 * 40];
  __shared__ __align__(16) _Float16 Bs[2][128 * 40];

  const int mode = (modeBase == 3) ? 3 : (int)blockIdx.z;
  const _Float16* Wt = (mode == 1) ? Wt1 : (mode == 2) ? Wt2 : Wt0;

  const int tid = threadIdx.x;
  const int lane = tid & 31, wv = tid >> 5;
  const int hi = lane >> 4, ln = lane & 15;
  const int wr = wv & 3, wc = wv >> 2;                 // 4 x 2 wave grid
  const int m0 = blockIdx.x * 128, n0 = blockIdx.y * 128;

  // each thread stages 2 A-chunks + 2 B-chunks of 8 halves per K-step
  const int rowA = tid >> 2, co = (tid & 3) * 8;       // rows rowA, rowA+64
  const _Float16* gA0 = A  + (size_t)(m0 + rowA)      * NDIM + co;
  const _Float16* gA1 = A  + (size_t)(m0 + rowA + 64) * NDIM + co;
  const _Float16* gB0 = Wt + (size_t)(n0 + rowA)      * NDIM + co;
  const _Float16* gB1 = Wt + (size_t)(n0 + rowA + 64) * NDIM + co;
  unsigned lA0[2], lA1[2], lB0[2], lB1[2];
  for (int s2 = 0; s2 < 2; ++s2) {
    lA0[s2] = lds_addr(&As[s2][rowA * 40 + co]);
    lA1[s2] = lds_addr(&As[s2][(rowA + 64) * 40 + co]);
    lB0[s2] = lds_addr(&Bs[s2][rowA * 40 + co]);
    lB1[s2] = lds_addr(&Bs[s2][(rowA + 64) * 40 + co]);
  }

  v8f acc[2][4] = {};

  // prologue: stage 0
  async_cp16(lA0[0], gA0); async_cp16(lA1[0], gA1);
  async_cp16(lB0[0], gB0); async_cp16(lB1[0], gB1);

  int cur = 0;
  for (int it = 0; it < NDIM / 32; ++it) {
    const int kk = it * 32;
    if (it + 1 < NDIM / 32) {
      const int nxt = cur ^ 1, kn = kk + 32;
      async_cp16(lA0[nxt], gA0 + kn); async_cp16(lA1[nxt], gA1 + kn);
      async_cp16(lB0[nxt], gB0 + kn); async_cp16(lB1[nxt], gB1 + kn);
      asm volatile("s_wait_asynccnt 0x4" ::: "memory");  // current stage landed
    } else {
      asm volatile("s_wait_asynccnt 0x0" ::: "memory");
    }
    __syncthreads();

    F16Frag a[2], b[4];
    for (int i = 0; i < 2; ++i) {
      const _Float16* p = &As[cur][((wr * 2 + i) * 16 + ln) * 40];
      a[i].u[0] = *(const uint4*)(p + 8 * hi);
      a[i].u[1] = *(const uint4*)(p + 16 + 8 * hi);
    }
    for (int j = 0; j < 4; ++j) {
      const _Float16* p = &Bs[cur][((wc * 4 + j) * 16 + ln) * 40 + 8 * hi];
      b[j].u[0] = *(const uint4*)(p);
      b[j].u[1] = *(const uint4*)(p + 16);
    }
    for (int i = 0; i < 2; ++i)
      for (int j = 0; j < 4; ++j)
        acc[i][j] = WMMA(a[i].v, b[j].v, acc[i][j]);
    __syncthreads();
    cur ^= 1;
  }

  for (int i = 0; i < 2; ++i)
    for (int j = 0; j < 4; ++j)
      for (int r = 0; r < 8; ++r) {
        int M = m0 + (wr * 2 + i) * 16 + r + 8 * hi;
        int N = n0 + (wc * 4 + j) * 16 + ln;
        float v = acc[i][j][r];
        if (mode == 3) {
          Dout[(size_t)M * NDIM + N] = v;
        } else {
          int bb = M >> 12, s = M & (NSEQ - 1);
          int hh = N >> 6, dh = N & 63;
          _Float16* dst = (mode == 0) ? D0 : (mode == 1) ? D1 : D2;
          float sv = (mode == 0) ? v * 0.125f : v;    // q / sqrt(64)
          dst[(((size_t)(bb * NHEAD + hh) * NSEQ) + s) * NDH + dh] = (_Float16)sv;
        }
      }
}

// ---------------------------- attention ------------------------------------
// one block per (b, h, chunk); 8 waves, each owns a 16-row query strip.
__global__ __launch_bounds__(256) void longformer_attn(
    const _Float16* __restrict__ qb, const _Float16* __restrict__ kb,
    const _Float16* __restrict__ vb, const int* __restrict__ gidx,
    _Float16* __restrict__ attn) {
  __shared__ __align__(16) _Float16 Qs[128 * 64];       // 16 KB
  __shared__ __align__(16) _Float16 Ks[384 * 64];       // 48 KB
  __shared__ __align__(16) _Float16 Vst[64 * 392];      // 49 KB (transposed, pad 8)
  __shared__ __align__(16) _Float16 Kg[64 * 64];        // 8 KB
  __shared__ __align__(16) _Float16 Vgt[64 * 72];       // 9 KB (transposed, pad 8)
  __shared__ __align__(16) _Float16 Ps[8 * 16 * 32];    // 8 KB prob bounce

  const int bid = blockIdx.x;
  const int c = bid & 31, h = (bid >> 5) & 15, b = bid >> 9;
  const int tid = threadIdx.x;
  const size_t base = (size_t)(b * NHEAD + h) * NSEQ * NDH;

  { // Q tile: straight async copy
    const uint4* src = (const uint4*)(qb + base + (size_t)c * 128 * NDH);
    uint4* dst = (uint4*)Qs;
    for (int u = tid; u < 1024; u += 256)
      async_cp16(lds_addr(&dst[u]), &src[u]);
  }
  // windowed K: async copy valid rows, zero-fill out-of-range rows
  for (int u = tid; u < 3072; u += 256) {
    int row = u >> 3, off = (u & 7) * 8;
    int pos = c * 128 - 128 + row;
    if (pos >= 0 && pos < NSEQ) {
      async_cp16(lds_addr(&Ks[row * 64 + off]),
                 kb + base + (size_t)pos * NDH + off);
    } else {
      uint4 z = {0, 0, 0, 0};
      *(uint4*)&Ks[row * 64 + off] = z;
    }
  }
  // windowed V transposed to [dh][key] (manual scatter)
  for (int u = tid; u < 3072; u += 256) {
    int row = u >> 3, off = (u & 7) * 8;
    int pos = c * 128 - 128 + row;
    uint4 vv = {0, 0, 0, 0};
    if (pos >= 0 && pos < NSEQ)
      vv = *(const uint4*)(vb + base + (size_t)pos * NDH + off);
    const _Float16* hv = (const _Float16*)&vv;
    for (int q = 0; q < 8; ++q) Vst[(off + q) * 392 + row] = hv[q];
  }
  // global K (async gather) + global V transposed (manual scatter)
  for (int u = tid; u < 512; u += 256) {
    int g = u >> 3, off = (u & 7) * 8;
    int pos = gidx[b * 64 + g];
    async_cp16(lds_addr(&Kg[g * 64 + off]),
               kb + base + (size_t)pos * NDH + off);
    uint4 vv = *(const uint4*)(vb + base + (size_t)pos * NDH + off);
    const _Float16* hv = (const _Float16*)&vv;
    for (int q = 0; q < 8; ++q) Vgt[(off + q) * 72 + g] = hv[q];
  }
  asm volatile("s_wait_asynccnt 0x0" ::: "memory");
  __syncthreads();

  const int lane = tid & 31, wv = tid >> 5;
  const int hi = lane >> 4, ln = lane & 15;

  F16Frag a0, a1;   // Q strip A-fragments, dh 0..31 and 32..63
  {
    const _Float16* qp = &Qs[(wv * 16 + ln) * 64];
    a0.u[0] = *(const uint4*)(qp + 8 * hi);
    a0.u[1] = *(const uint4*)(qp + 16 + 8 * hi);
    a1.u[0] = *(const uint4*)(qp + 32 + 8 * hi);
    a1.u[1] = *(const uint4*)(qp + 48 + 8 * hi);
  }

  float m[8], l[8];
  for (int r = 0; r < 8; ++r) { m[r] = -3.0e38f; l[r] = 0.f; }

  // pass 1: online max/sum over 28 key tiles (4 global + 24 windowed)
  for (int t = 0; t < 28; ++t) {
    const _Float16* kp = (t < 4) ? &Kg[(t * 16 + ln) * 64]
                                 : &Ks[((t - 4) * 16 + ln) * 64];
    F16Frag b0, b1;
    b0.u[0] = *(const uint4*)(kp + 8 * hi);
    b0.u[1] = *(const uint4*)(kp + 16 + 8 * hi);
    b1.u[0] = *(const uint4*)(kp + 32 + 8 * hi);
    b1.u[1] = *(const uint4*)(kp + 48 + 8 * hi);
    v8f sacc = {};
    sacc = WMMA(a0.v, b0.v, sacc);
    sacc = WMMA(a1.v, b1.v, sacc);
    for (int r = 0; r < 8; ++r) {
      float sv = sacc[r];
      if (t >= 4) {
        int qr = wv * 16 + r + 8 * hi;          // query row in chunk
        int j = (t - 4) * 16 + ln;              // window key index
        int pos = c * 128 - 128 + j;
        bool ok = (j >= qr) && (j <= qr + 256) && (pos >= 0) && (pos < NSEQ);
        sv = ok ? sv : NEGV;
      }
      float mt = sv;
      mt = fmaxf(mt, __shfl_xor(mt, 1, 32));
      mt = fmaxf(mt, __shfl_xor(mt, 2, 32));
      mt = fmaxf(mt, __shfl_xor(mt, 4, 32));
      mt = fmaxf(mt, __shfl_xor(mt, 8, 32));
      float mn = fmaxf(m[r], mt);
      float p = __expf(sv - mn);
      p += __shfl_xor(p, 1, 32);
      p += __shfl_xor(p, 2, 32);
      p += __shfl_xor(p, 4, 32);
      p += __shfl_xor(p, 8, 32);
      l[r] = l[r] * __expf(m[r] - mn) + p;
      m[r] = mn;
    }
  }

  // pass 2: recompute scores, probs -> LDS bounce -> P @ V via WMMA
  v8f oacc[4] = {};
  for (int tp = 0; tp < 14; ++tp) {
    for (int u2 = 0; u2 < 2; ++u2) {
      int t = tp * 2 + u2;
      const _Float16* kp = (t < 4) ? &Kg[(t * 16 + ln) * 64]
                                   : &Ks[((t - 4) * 16 + ln) * 64];
      F16Frag b0, b1;
      b0.u[0] = *(const uint4*)(kp + 8 * hi);
      b0.u[1] = *(const uint4*)(kp + 16 + 8 * hi);
      b1.u[0] = *(const uint4*)(kp + 32 + 8 * hi);
      b1.u[1] = *(const uint4*)(kp + 48 + 8 * hi);
      v8f sacc = {};
      sacc = WMMA(a0.v, b0.v, sacc);
      sacc = WMMA(a1.v, b1.v, sacc);
      for (int r = 0; r < 8; ++r) {
        float sv = sacc[r];
        if (t >= 4) {
          int qr = wv * 16 + r + 8 * hi;
          int j = (t - 4) * 16 + ln;
          int pos = c * 128 - 128 + j;
          bool ok = (j >= qr) && (j <= qr + 256) && (pos >= 0) && (pos < NSEQ);
          sv = ok ? sv : NEGV;
        }
        float p = __expf(sv - m[r]);            // exp(NEG - m) underflows to 0
        Ps[wv * 512 + (r + 8 * hi) * 32 + u2 * 16 + ln] = (_Float16)p;
      }
    }
    __asm__ volatile("s_wait_dscnt 0" ::: "memory");  // cross-lane LDS hand-off
    F16Frag pa;
    {
      const _Float16* pp = &Ps[wv * 512 + ln * 32];
      pa.u[0] = *(const uint4*)(pp + 8 * hi);
      pa.u[1] = *(const uint4*)(pp + 16 + 8 * hi);
    }
    const _Float16* vsrc; int vstride, kbase;
    if (tp < 2) { vsrc = Vgt; vstride = 72;  kbase = tp * 32; }
    else        { vsrc = Vst; vstride = 392; kbase = (tp - 2) * 32; }
    for (int j = 0; j < 4; ++j) {
      const _Float16* vp = vsrc + (j * 16 + ln) * vstride + kbase + 8 * hi;
      F16Frag vbf;
      vbf.u[0] = *(const uint4*)(vp);
      vbf.u[1] = *(const uint4*)(vp + 16);
      oacc[j] = WMMA(pa.v, vbf.v, oacc[j]);
    }
    __asm__ volatile("" ::: "memory");
  }

  // normalize and store f16 attn output at [b][s][h*64+dh]
  for (int r = 0; r < 8; ++r) {
    float inv = 1.0f / l[r];
    int srow = c * 128 + wv * 16 + r + 8 * hi;
    size_t rowBase = ((size_t)(b * NSEQ + srow)) * NDIM + h * NDH;
    for (int j = 0; j < 4; ++j) {
      int dh = j * 16 + ln;
      attn[rowBase + dh] = (_Float16)(oacc[j][r] * inv);
    }
  }
}

// ------------------------------ launcher -----------------------------------

extern "C" void kernel_launch(void* const* d_in, const int* in_sizes, int n_in,
                              void* d_out, int out_size, void* d_ws, size_t ws_size,
                              hipStream_t stream) {
  const float* x  = (const float*)d_in[0];
  const float* Wq = (const float*)d_in[1];
  const float* Wk = (const float*)d_in[2];
  const float* Wv = (const float*)d_in[3];
  const float* Wo = (const float*)d_in[4];
  const int* gidx = (const int*)d_in[5];

  char* ws = (char*)d_ws;
  size_t off = 0;
  auto alloc = [&](size_t bytes) -> char* {
    char* p = ws + off;
    off += (bytes + 255) & ~(size_t)255;
    return p;
  };
  const size_t NTOK = (size_t)2 * NSEQ;                 // 8192 rows
  _Float16* xh   = (_Float16*)alloc(NTOK * NDIM * 2);
  _Float16* WqT  = (_Float16*)alloc((size_t)NDIM * NDIM * 2);
  _Float16* WkT  = (_Float16*)alloc((size_t)NDIM * NDIM * 2);
  _Float16* WvT  = (_Float16*)alloc((size_t)NDIM * NDIM * 2);
  _Float16* WoT  = (_Float16*)alloc((size_t)NDIM * NDIM * 2);
  _Float16* qbuf = (_Float16*)alloc(NTOK * NDIM * 2);
  _Float16* kbuf = (_Float16*)alloc(NTOK * NDIM * 2);
  _Float16* vbuf = (_Float16*)alloc(NTOK * NDIM * 2);
  _Float16* attn = (_Float16*)alloc(NTOK * NDIM * 2);
  (void)ws_size; (void)in_sizes; (void)n_in; (void)out_size;

  int n4 = (int)(NTOK * NDIM / 4);
  cvt_f32_to_f16<<<(n4 + 255) / 256, 256, 0, stream>>>(x, xh, n4);
  cvt_transpose<<<dim3(32, 32), 256, 0, stream>>>(Wq, WqT);
  cvt_transpose<<<dim3(32, 32), 256, 0, stream>>>(Wk, WkT);
  cvt_transpose<<<dim3(32, 32), 256, 0, stream>>>(Wv, WvT);
  cvt_transpose<<<dim3(32, 32), 256, 0, stream>>>(Wo, WoT);

  // Q, K, V projections (grid.z selects weight/output)
  gemm128<<<dim3(64, 8, 3), 256, 0, stream>>>(xh, WqT, WkT, WvT,
                                              qbuf, kbuf, vbuf, nullptr, 0);
  // attention: one block per (b, h, chunk)
  longformer_attn<<<dim3(2 * NHEAD * 32), 256, 0, stream>>>(qbuf, kbuf, vbuf,
                                                            gidx, attn);
  // output projection, f32 result
  gemm128<<<dim3(64, 8, 1), 256, 0, stream>>>(attn, WoT, WoT, WoT,
                                              nullptr, nullptr, nullptr,
                                              (float*)d_out, 3);
}
